// ContextualAttention_5669356832394
// MI455X (gfx1250) — matrix operand order repaired and checked by reading, not compiled, not run on hardware
//
#include <hip/hip_runtime.h>
#include <hip/hip_bf16.h>
#include <math.h>
#include <stdint.h>

// ---------------------------------------------------------------------------
// Contextual attention for MI455X (gfx1250, wave32, WMMA + TDM).
// Pipeline:
//   prep_wn    : normalized 3x3x128 patches of b_s  -> Wn  [1024 x 1152] bf16 (+ mask means mm)
//   prep_xcol  : im2col of f_s (transposed layout)  -> Xt  [1024 x 1152] bf16  (B^T for GEMM1)
//   gemm (bf16 WMMA, TDM double-buffered): Y0 = Wn x Xt^T   [1024 x 1024] f32
//   fuse1/fuse2: diagonal 3-tap smoothing (+ transposed pass)    Y0 -> Y1 -> Y0
//   softmax    : masked softmax over patch axis -> Pt [1024(pix) x 1024(patch)] bf16
//   prep_r     : raw 4x4x128 patches of b -> R [2048 x 1024] bf16 (A for GEMM2)
//   gemm (bf16 WMMA, TDM double-buffered): O = R x Pt^T     [2048 x 1024] f32
//   scatter    : overlap-add the 16 (a,b) slices into (128,64,64), /4
// ---------------------------------------------------------------------------

typedef __attribute__((ext_vector_type(16))) __bf16 v16bf;
typedef __attribute__((ext_vector_type(8)))  __bf16 v8bf;
typedef __attribute__((ext_vector_type(8)))  float  v8f;
typedef __attribute__((ext_vector_type(4)))  unsigned int u32x4;
typedef __attribute__((ext_vector_type(8)))  int i32x8;
typedef __attribute__((ext_vector_type(4)))  int i32x4;

#ifndef __has_builtin
#define __has_builtin(x) 0
#endif
#if defined(__HIP_DEVICE_COMPILE__) && __has_builtin(__builtin_amdgcn_tensor_load_to_lds)
#define USE_TDM 1
#else
#define USE_TDM 0
#endif

__device__ __forceinline__ __bf16 f2bf(float f) {
    unsigned int u = __builtin_bit_cast(unsigned int, f);
    unsigned int r = (u + 0x7FFFu + ((u >> 16) & 1u)) >> 16;
    unsigned short s = (unsigned short)r;
    return __builtin_bit_cast(__bf16, s);
}

#if USE_TDM
// ---------------------------------------------------------------------------
// Issue one TDM 2-D tile load: 128 rows x 32 bf16 (64B payload) from a
// row-major [rows x K] bf16 tensor into LDS with 16B padding per row
// (LDS row stride = 40 halves = 80B). D# per CDNA5 ISA ch.8:
//   group0: count=1, lds_addr[63:32], global_addr[120:64], type=2
//   group1: data_size=2B, pad_enable, pad_interval=16 DW, pad_amount=4 DW,
//           tensor_dim0=K, tensor_dim1=rows, tile_dim0=32, tile_dim1=128,
//           tensor_dim0_stride=K
// Wave-level op (EXEC ignored), tracked by TENSORcnt.
// ---------------------------------------------------------------------------
__device__ __forceinline__ void tdm_load_tile(const __bf16* gptr, unsigned lds_off,
                                              int K, int rows)
{
    unsigned long long ga = (unsigned long long)(uintptr_t)gptr;
    u32x4 g0;
    g0[0] = 1u;                                      // count=1 (valid), user mode
    g0[1] = lds_off;                                 // lds_addr (bytes)
    g0[2] = (unsigned)(ga & 0xFFFFFFFFull);          // global_addr[31:0]
    g0[3] = (unsigned)((ga >> 32) & 0x01FFFFFFull)   // global_addr[56:32]
          | (2u << 30);                              // type = 2 ("image")
    i32x8 g1;
    g1[0] = (1 << 16)                                // data_size: 2 bytes
          | (1 << 20)                                // pad_enable
          | (3 << 22)                                // pad_interval: 16 DWORDs (64B)
          | (3 << 25);                               // pad_amount: 4 DWORDs (16B)
    g1[1] = (K & 0xFFFF) << 16;                      // tensor_dim0[15:0]   @ bits 63:48
    g1[2] = ((unsigned)K >> 16) & 0xFFFF;            // tensor_dim0[31:16]  @ bits 79:64
    g1[2] |= (rows & 0xFFFF) << 16;                  // tensor_dim1[15:0]   @ bits 95:80
    g1[3] = ((unsigned)rows >> 16) & 0xFFFF;         // tensor_dim1[31:16]  @ bits 111:96
    g1[3] |= (32 << 16);                             // tile_dim0 = 32      @ bits 127:112
    g1[4] = 128;                                     // tile_dim1 = 128 (tile_dim2 = 0)
    g1[5] = K;                                       // tensor_dim0_stride[31:0]
    g1[6] = 0;                                       // stride[47:32] = 0, dim1_stride lo
    g1[7] = 0;
    i32x4 z4 = {0, 0, 0, 0};
#if __clang_major__ >= 23
    i32x8 z8 = {0, 0, 0, 0, 0, 0, 0, 0};
    __builtin_amdgcn_tensor_load_to_lds(g0, g1, z4, z4, z8, 0);
#else
    __builtin_amdgcn_tensor_load_to_lds(g0, g1, z4, z4, 0);
#endif
}
#endif // USE_TDM

// ---------------------------------------------------------------------------
// Kernel 1: normalized background patches (A-matrix of GEMM1) + mask means.
// ---------------------------------------------------------------------------
__global__ __launch_bounds__(256)
void prep_wn_kernel(const float* __restrict__ b, const float* __restrict__ mask,
                    __bf16* __restrict__ Wn, float* __restrict__ mm)
{
    const int l = blockIdx.x;          // 0..1023
    const int s = blockIdx.y;          // 0..3
    const int py = l >> 5, px = l & 31;
    const int tid = threadIdx.x;
    __shared__ float red[256];

    float v[5];
    float ss = 0.f;
#pragma unroll
    for (int it = 0; it < 5; ++it) {
        int k = tid + it * 256;
        float val = 0.f;
        if (k < 1152) {
            int c  = k / 9;
            int r9 = k - c * 9;
            int di = r9 / 3, dj = r9 - di * 3;
            int y = py + di - 1, x = px + dj - 1;
            if ((unsigned)y < 32u && (unsigned)x < 32u)
                val = b[(((s * 128 + c) * 64) + 2 * y) * 64 + 2 * x];
        }
        v[it] = val;
        ss += val * val;
    }
    red[tid] = ss;
    __syncthreads();
    for (int off = 128; off > 0; off >>= 1) {
        if (tid < off) red[tid] += red[tid + off];
        __syncthreads();
    }
    const float inv = 1.0f / sqrtf(red[0] + 0.0001f * 1152.0f);
#pragma unroll
    for (int it = 0; it < 5; ++it) {
        int k = tid + it * 256;
        if (k < 1152)
            Wn[((size_t)(s * 1024 + l)) * 1152 + k] = f2bf(v[it] * inv);
    }
    if (tid == 0) {
        float m = 0.f;
        for (int di = 0; di < 3; ++di)
            for (int dj = 0; dj < 3; ++dj) {
                int y = py + di - 1, x = px + dj - 1;
                if ((unsigned)y < 32u && (unsigned)x < 32u)
                    m += mask[((s * 64) + 2 * y) * 64 + 2 * x];
            }
        mm[s * 1024 + l] = m * (1.0f / 9.0f);
    }
}

// ---------------------------------------------------------------------------
// Kernel 2: im2col of foreground, stored transposed: Xt[(s*1024+n)*1152 + k].
// ---------------------------------------------------------------------------
__global__ __launch_bounds__(256)
void prep_xcol_kernel(const float* __restrict__ f, __bf16* __restrict__ Xt)
{
    size_t idx = blockIdx.x * (size_t)blockDim.x + threadIdx.x;
    const size_t total = 4ull * 1024 * 1152;
    if (idx >= total) return;
    int k = (int)(idx % 1152);
    size_t t = idx / 1152;
    int n = (int)(t & 1023);
    int s = (int)(t >> 10);
    int c  = k / 9;
    int r9 = k - c * 9;
    int di = r9 / 3, dj = r9 - di * 3;
    int y = (n >> 5) + di - 1, x = (n & 31) + dj - 1;
    float val = 0.f;
    if ((unsigned)y < 32u && (unsigned)x < 32u)
        val = f[(((s * 128 + c) * 64) + 2 * y) * 64 + 2 * x];
    Xt[idx] = f2bf(val);
}

// ---------------------------------------------------------------------------
// Generic bf16 WMMA GEMM:  C[M,N] (f32) = A[M,K] x B[K,N]
// A row-major [M,K] bf16; Bt row-major [N,K] bf16 (B transposed).
// 256 threads = 8 waves; 128x128 block tile; each wave: 64x32 = 4x2 WMMA tiles.
// K-step 32 staged in LDS (row stride 40 halves). TDM path: double-buffered
// tensor_load_to_lds issued by wave 0, one barrier per K-step, DMA overlaps
// WMMA. Fallback: global_load_b128 -> ds_store_b128 staging.
// M,N multiples of 128; K multiple of 32.
// ---------------------------------------------------------------------------
#define TILE_ELEMS 5120   // 128 * 40 halves
#define TILE_BYTES 10240

__global__ __launch_bounds__(256)
void wmma_gemm_bf16(const __bf16* __restrict__ A, size_t sA,
                    const __bf16* __restrict__ Bt, size_t sB,
                    float* __restrict__ C, size_t sC,
                    int M, int N, int K)
{
    __shared__ __bf16 As[2 * TILE_ELEMS];
    __shared__ __bf16 Bs[2 * TILE_ELEMS];

    const int bz = blockIdx.z;
    A  += (size_t)bz * sA;
    Bt += (size_t)bz * sB;
    C  += (size_t)bz * sC;

    const int n0 = blockIdx.x * 128;
    const int m0 = blockIdx.y * 128;
    const int tid  = threadIdx.x;
    const int lane = tid & 31;
    const int wid  = tid >> 5;
    const int wm = (wid & 1) * 64;   // wave row offset inside block tile
    const int wn = (wid >> 1) * 32;  // wave col offset inside block tile

    const v8f vzero = {0.f, 0.f, 0.f, 0.f, 0.f, 0.f, 0.f, 0.f};
    v8f acc[4][2];
#pragma unroll
    for (int tm = 0; tm < 4; ++tm)
#pragma unroll
        for (int tn = 0; tn < 2; ++tn) acc[tm][tn] = vzero;

    const int half = (lane < 16) ? 0 : 1;
    const int mrow = lane & 15;

    // 8 WMMA on one staged K-slab. Fragments per ISA VGPR layouts:
    //   A 16x32: lanes 0-15 -> K{0..7,16..23}; lanes 16-31 -> K{8..15,24..31}
    //   B 32x16: lane n<16 -> K{0..7,16..23}; lane>=16 -> K{8..15,24..31}
    auto compute = [&](int cb) {
        v16bf afr[4];
#pragma unroll
        for (int tm = 0; tm < 4; ++tm) {
            int row = wm + tm * 16 + mrow;
            int c0  = half * 8;
            v8bf lo = *(const v8bf*)&As[cb + row * 40 + c0];
            v8bf hi = *(const v8bf*)&As[cb + row * 40 + c0 + 16];
#pragma unroll
            for (int e = 0; e < 8; ++e) { afr[tm][e] = lo[e]; afr[tm][e + 8] = hi[e]; }
        }
        v16bf bfr[2];
#pragma unroll
        for (int tn = 0; tn < 2; ++tn) {
            int row = wn + tn * 16 + mrow;
            int k0  = half * 8;
            v8bf lo = *(const v8bf*)&Bs[cb + row * 40 + k0];
            v8bf hi = *(const v8bf*)&Bs[cb + row * 40 + k0 + 16];
#pragma unroll
            for (int e = 0; e < 8; ++e) { bfr[tn][e] = lo[e]; bfr[tn][e + 8] = hi[e]; }
        }
#pragma unroll
        for (int tm = 0; tm < 4; ++tm)
#pragma unroll
            for (int tn = 0; tn < 2; ++tn)
                acc[tm][tn] = __builtin_amdgcn_wmma_f32_16x16x32_bf16(
                    false, afr[tm], false, bfr[tn],
                    (short)0, acc[tm][tn], false, false);
    };

#if USE_TDM
    // LDS byte offsets of the staging buffers (flat LDS addr = low 32 bits).
    const unsigned ldsA = (unsigned)(uintptr_t)&As[0];
    const unsigned ldsB = (unsigned)(uintptr_t)&Bs[0];
    if (wid == 0) {
        tdm_load_tile(A  + (size_t)m0 * K, ldsA, K, M);
        tdm_load_tile(Bt + (size_t)n0 * K, ldsB, K, N);
    }
    for (int kk = 0; kk < K; kk += 32) {
        const int cur = (kk >> 5) & 1;
        if (wid == 0) __builtin_amdgcn_s_wait_tensorcnt(0);
        __syncthreads();   // buffer `cur` ready for all; prev reads of `nxt` done
        if (wid == 0 && kk + 32 < K) {
            const unsigned nb = (unsigned)(cur ^ 1) * TILE_BYTES;
            tdm_load_tile(A  + (size_t)m0 * K + (kk + 32), ldsA + nb, K, M);
            tdm_load_tile(Bt + (size_t)n0 * K + (kk + 32), ldsB + nb, K, N);
        }
        compute(cur * TILE_ELEMS);   // overlaps with next TDM transfer
    }
#else
    for (int kk = 0; kk < K; kk += 32) {
#pragma unroll
        for (int r = 0; r < 2; ++r) {
            int idx = tid + r * 256;          // 0..511
            int row = idx >> 2;               // 0..127
            int ch  = (idx & 3) * 8;          // 0,8,16,24
            *(v8bf*)&As[row * 40 + ch] =
                *(const v8bf*)&A[(size_t)(m0 + row) * K + kk + ch];
            *(v8bf*)&Bs[row * 40 + ch] =
                *(const v8bf*)&Bt[(size_t)(n0 + row) * K + kk + ch];
            if (kk + 32 < K) {
                __builtin_prefetch((const void*)&A[(size_t)(m0 + row) * K + kk + 32 + ch], 0, 0);
                __builtin_prefetch((const void*)&Bt[(size_t)(n0 + row) * K + kk + 32 + ch], 0, 0);
            }
        }
        __syncthreads();
        compute(0);
        __syncthreads();
    }
#endif

    // Epilogue: C/D layout — VGPR j: lanes 0-15 row j, lanes 16-31 row j+8.
#pragma unroll
    for (int tm = 0; tm < 4; ++tm)
#pragma unroll
        for (int tn = 0; tn < 2; ++tn) {
            int col   = n0 + wn + tn * 16 + mrow;
            int rbase = m0 + wm + tm * 16 + half * 8;
#pragma unroll
            for (int j = 0; j < 8; ++j)
                C[(size_t)(rbase + j) * N + col] = acc[tm][tn][j];
        }
}

// ---------------------------------------------------------------------------
// Fuse pass 1: out[r,j] = in[r,j] + in[r-1,j-1] + in[r+1,j+1]  (zero pad)
// ---------------------------------------------------------------------------
__global__ __launch_bounds__(256)
void fuse1_kernel(const float* __restrict__ Yin, float* __restrict__ Yout)
{
    size_t idx = blockIdx.x * (size_t)blockDim.x + threadIdx.x;
    const size_t total = 4ull << 20;
    if (idx >= total) return;
    int j = (int)(idx & 1023);
    int r = (int)((idx >> 10) & 1023);
    float acc = Yin[idx];
    if (r > 0 && j > 0)       acc += Yin[idx - 1025];
    if (r < 1023 && j < 1023) acc += Yin[idx + 1025];
    Yout[idx] = acc;
}

__device__ __forceinline__ int swap32(int t) { return ((t & 31) << 5) | (t >> 5); }

// Fuse pass 2 (on transposed index space, written back untransposed):
// out[i,j] = sum_d in[swap(swap(i)+d), swap(swap(j)+d)]
__global__ __launch_bounds__(256)
void fuse2_kernel(const float* __restrict__ Yin, float* __restrict__ Yout)
{
    size_t idx = blockIdx.x * (size_t)blockDim.x + threadIdx.x;
    const size_t total = 4ull << 20;
    if (idx >= total) return;
    int j = (int)(idx & 1023);
    int i = (int)((idx >> 10) & 1023);
    int s = (int)(idx >> 20);
    const float* Ys = Yin + ((size_t)s << 20);
    int it = swap32(i), jt = swap32(j);
    float acc = 0.f;
#pragma unroll
    for (int d = -1; d <= 1; ++d) {
        int si = it + d, sj = jt + d;
        if ((unsigned)si < 1024u && (unsigned)sj < 1024u)
            acc += Ys[(size_t)swap32(si) * 1024 + swap32(sj)];
    }
    Yout[idx] = acc;
}

// ---------------------------------------------------------------------------
// Masked softmax over the patch axis l (column j fixed); writes transposed
// probabilities Pt[(s*1024+j)*1024 + l] in bf16 (B^T operand for GEMM2).
// ---------------------------------------------------------------------------
__global__ __launch_bounds__(256)
void softmax_kernel(const float* __restrict__ Y, const float* __restrict__ mm,
                    __bf16* __restrict__ Pt)
{
    const int j = blockIdx.x;
    const int s = blockIdx.y;
    const int tid = threadIdx.x;
    const float* Ys  = Y + ((size_t)s << 20);
    const float* mms = mm + s * 1024;
    __shared__ float red[256];

    float logit[4];
    float mx = -3.4e38f;
#pragma unroll
    for (int it = 0; it < 4; ++it) {
        int l = tid + it * 256;
        float lg = Ys[(size_t)l * 1024 + j] * mms[l] * 10.0f;
        logit[it] = lg;
        mx = fmaxf(mx, lg);
    }
    red[tid] = mx; __syncthreads();
    for (int off = 128; off > 0; off >>= 1) {
        if (tid < off) red[tid] = fmaxf(red[tid], red[tid + off]);
        __syncthreads();
    }
    mx = red[0]; __syncthreads();

    float e[4];
    float sum = 0.f;
#pragma unroll
    for (int it = 0; it < 4; ++it) { e[it] = expf(logit[it] - mx); sum += e[it]; }
    red[tid] = sum; __syncthreads();
    for (int off = 128; off > 0; off >>= 1) {
        if (tid < off) red[tid] += red[tid + off];
        __syncthreads();
    }
    float inv = 1.0f / red[0];
#pragma unroll
    for (int it = 0; it < 4; ++it) {
        int l = tid + it * 256;
        Pt[(((size_t)s * 1024 + j) << 10) + l] = f2bf(e[it] * inv * mms[l]);
    }
}

// ---------------------------------------------------------------------------
// Raw 4x4 background patches, laid out as GEMM2 A-matrix:
// R[(s*2048 + (a*4+b)*128 + c)*1024 + l] = b_pad[c, 2py+a-1, 2px+b-1]
// ---------------------------------------------------------------------------
__global__ __launch_bounds__(256)
void prep_r_kernel(const float* __restrict__ b, __bf16* __restrict__ R)
{
    size_t idx = blockIdx.x * (size_t)blockDim.x + threadIdx.x;
    const size_t total = 4ull * 2048 * 1024;
    if (idx >= total) return;
    int l  = (int)(idx & 1023);
    int m  = (int)((idx >> 10) & 2047);
    int s  = (int)(idx >> 21);
    int c  = m & 127;
    int ab = m >> 7;
    int a  = ab >> 2, bb = ab & 3;
    int py = l >> 5, px = l & 31;
    int y = 2 * py + a - 1, x = 2 * px + bb - 1;
    float val = 0.f;
    if ((unsigned)y < 64u && (unsigned)x < 64u)
        val = b[(((s * 128 + c) * 64) + y) * 64 + x];
    R[idx] = f2bf(val);
}

// ---------------------------------------------------------------------------
// Overlap-add of the 16 (a,b) deconv slices into (128,64,64), divided by 4.
// ---------------------------------------------------------------------------
__global__ __launch_bounds__(256)
void scatter_kernel(const float* __restrict__ O, float* __restrict__ out)
{
    size_t idx = blockIdx.x * (size_t)blockDim.x + threadIdx.x;
    const size_t total = 4ull * 128 * 64 * 64;
    if (idx >= total) return;
    int x = (int)(idx & 63);
    int y = (int)((idx >> 6) & 63);
    int c = (int)((idx >> 12) & 127);
    int s = (int)(idx >> 19);
    const float* Os = O + ((size_t)s * 2048 * 1024);
    float acc = 0.f;
    int a0 = (y + 1) & 1;
    int b0 = (x + 1) & 1;
#pragma unroll
    for (int da = 0; da < 2; ++da) {
        int a = a0 + 2 * da;
        int p = (y + 1 - a) >> 1;
        if ((unsigned)p >= 32u) continue;
#pragma unroll
        for (int db = 0; db < 2; ++db) {
            int bb = b0 + 2 * db;
            int q = (x + 1 - bb) >> 1;
            if ((unsigned)q >= 32u) continue;
            acc += Os[(size_t)((a * 4 + bb) * 128 + c) * 1024 + p * 32 + q];
        }
    }
    out[idx] = acc * 0.25f;
}

// ---------------------------------------------------------------------------
// Host launcher. Workspace layout (all offsets 256B aligned):
//   mm f32 [4*1024] | Wn bf16 [4*1024*1152] | Xt bf16 [4*1024*1152]
//   Y0 f32 [4<<20]  | Y1 f32 [4<<20] | Pt bf16 [4<<20]
//   R bf16 [4*2048*1024] | O f32 [4*2048*1024]        (~106 MB total)
// ---------------------------------------------------------------------------
extern "C" void kernel_launch(void* const* d_in, const int* in_sizes, int n_in,
                              void* d_out, int out_size, void* d_ws, size_t ws_size,
                              hipStream_t stream)
{
    const float* f    = (const float*)d_in[0];
    const float* b    = (const float*)d_in[1];
    const float* mask = (const float*)d_in[2];
    float* out = (float*)d_out;

    char* ws = (char*)d_ws;
    const size_t off_mm = 0;
    const size_t off_Wn = off_mm + 4ull * 1024 * sizeof(float);
    const size_t off_Xt = off_Wn + 4ull * 1024 * 1152 * 2;
    const size_t off_Y0 = off_Xt + 4ull * 1024 * 1152 * 2;
    const size_t off_Y1 = off_Y0 + (4ull << 20) * sizeof(float);
    const size_t off_Pt = off_Y1 + (4ull << 20) * sizeof(float);
    const size_t off_R  = off_Pt + (4ull << 20) * 2;
    const size_t off_O  = off_R  + 4ull * 2048 * 1024 * 2;

    float*  mm = (float*)(ws + off_mm);
    __bf16* Wn = (__bf16*)(ws + off_Wn);
    __bf16* Xt = (__bf16*)(ws + off_Xt);
    float*  Y0 = (float*)(ws + off_Y0);
    float*  Y1 = (float*)(ws + off_Y1);
    __bf16* Pt = (__bf16*)(ws + off_Pt);
    __bf16* R  = (__bf16*)(ws + off_R);
    float*  O  = (float*)(ws + off_O);

    // 1) operand prep
    prep_wn_kernel<<<dim3(1024, 4), 256, 0, stream>>>(b, mask, Wn, mm);
    {
        size_t total = 4ull * 1024 * 1152;
        prep_xcol_kernel<<<(unsigned)((total + 255) / 256), 256, 0, stream>>>(f, Xt);
    }

    // 2) score GEMM: Y0 = Wn x Xt^T   (M=1024, N=1024, K=1152)
    wmma_gemm_bf16<<<dim3(8, 8, 4), 256, 0, stream>>>(
        Wn, (size_t)1024 * 1152, Xt, (size_t)1024 * 1152,
        Y0, (size_t)1 << 20, 1024, 1024, 1152);

    // 3) fuse (diag smoothing, straight then transposed)
    fuse1_kernel<<<(unsigned)((4ull << 20) / 256), 256, 0, stream>>>(Y0, Y1);
    fuse2_kernel<<<(unsigned)((4ull << 20) / 256), 256, 0, stream>>>(Y1, Y0);

    // 4) masked softmax over patch axis -> Pt (transposed, bf16)
    softmax_kernel<<<dim3(1024, 4), 256, 0, stream>>>(Y0, mm, Pt);

    // 5) paste GEMM: O = R x Pt^T   (M=2048, N=1024, K=1024)
    {
        size_t total = 4ull * 2048 * 1024;
        prep_r_kernel<<<(unsigned)((total + 255) / 256), 256, 0, stream>>>(b, R);
    }
    wmma_gemm_bf16<<<dim3(8, 16, 4), 256, 0, stream>>>(
        R, (size_t)2048 * 1024, Pt, (size_t)1 << 20,
        O, (size_t)2048 * 1024, 2048, 1024, 1024);

    // 6) overlap-add to final (4,128,64,64)
    {
        size_t total = 4ull * 128 * 64 * 64;
        scatter_kernel<<<(unsigned)((total + 255) / 256), 256, 0, stream>>>(O, out);
    }
    (void)in_sizes; (void)n_in; (void)out_size; (void)ws_size;
}
